// RouteNet_27264452395697
// MI455X (gfx1250) — matrix operand order repaired and checked by reading, not compiled, not run on hardware
//
#include <hip/hip_runtime.h>
#include <hip/hip_bf16.h>
#include <math.h>

typedef __attribute__((ext_vector_type(16))) _Float16 v16h;
typedef __attribute__((ext_vector_type(8)))  float    v8f;

#define NLINKS 8192
#define NPATHS 32768
#define PLEN   8
#define DD     32
#define TITER  8
#define NW_P   2   // waves per block, path kernel
#define NW_L   4   // waves per block, link kernel

// ---------------------------------------------------------------- fast math

__device__ __forceinline__ float fast_rcp(float x) { return __builtin_amdgcn_rcpf(x); }
__device__ __forceinline__ float fast_exp(float x) {
  return __builtin_amdgcn_exp2f(x * 1.4426950408889634f);   // v_exp_f32 is exp2
}

#if defined(__has_builtin)
#if __has_builtin(__builtin_amdgcn_tanhf)
#define HAVE_HW_TANH 1
#endif
#endif

__device__ __forceinline__ float fast_tanh(float x) {
#ifdef HAVE_HW_TANH
  return __builtin_amdgcn_tanhf(x);                          // v_tanh_f32 (gfx1250)
#else
  float t = __builtin_amdgcn_exp2f(x * 2.8853900817779268f); // exp(2x)
  return 1.0f - 2.0f * fast_rcp(t + 1.0f);
#endif
}

__device__ __forceinline__ float fast_sigmoid(float x) {
  return fast_rcp(1.0f + __builtin_amdgcn_exp2f(-1.4426950408889634f * x));
}

__device__ __forceinline__ float seluf_(float x) {
  const float sc = 1.0507009873554805f, al = 1.6732632423543772f;
  return x > 0.0f ? sc * x : sc * al * (expf(x) - 1.0f);
}

// ---------------------------------------------------------------- WMMA helpers

__device__ __forceinline__ v8f wmma16(v16h a, v16h b, v8f c) {
  // D(16x16,f32) = A(16x32,f16) x B(32x16,f16) + C
  return __builtin_amdgcn_wmma_f32_16x16x32_f16(
      /*neg_a=*/false, a, /*neg_b=*/false, b,
      /*c_mod=*/(short)0, c, /*reuse_a=*/false, /*reuse_b=*/false);
}

// B fragment (32x16 f16) from a [32 x 96] row-major weight matrix, col tile n0.
// ISA B layout (table order, matching A): lane holds column n = n0 + (lane&15);
//   element i -> K = (i&7) + 16*(i>>3) + 8*(lane>>4)
__device__ __forceinline__ v16h load_bfrag(const float* __restrict__ W,
                                           int n0, int m, int half) {
  v16h b;
#pragma unroll
  for (int i = 0; i < 16; ++i) {
    int k = (i & 7) + ((i >> 3) << 4) + (half << 3);
    b[i] = (_Float16)W[k * 96 + n0 + m];
  }
  return b;
}

// A fragment (16x32 f16): explicit ISA table layout. Lane's row = lane&15,
// element i holds K = (i&7) + 16*(i>>3) + 8*(lane>>4)  -> two contiguous 8-float
// runs of the source row: [8*half, +8) and [16+8*half, +8).
__device__ __forceinline__ v16h load_afrag_g(const float* __restrict__ row,
                                             int half, float s) {
  v16h a;
  const float* q0 = row + (half << 3);
  const float* q1 = row + 16 + (half << 3);
#pragma unroll
  for (int i = 0; i < 8; ++i) {
    a[i]     = (_Float16)(q0[i] * s);
    a[8 + i] = (_Float16)(q1[i] * s);
  }
  return a;
}

// A fragment from an LDS staging buffer laid out [16][33] (row = path, col = feature)
__device__ __forceinline__ v16h load_afrag_lds(const float* stage, int m, int half) {
  v16h a;
  const float* row = stage + m * 33;
#pragma unroll
  for (int i = 0; i < 8; ++i) {
    a[i]     = (_Float16)row[(half << 3) + i];
    a[8 + i] = (_Float16)row[16 + (half << 3) + i];
  }
  return a;
}

// ---------------------------------------------------------------- init / zero

__global__ void init_kernel(const float* __restrict__ cap,
                            const float* __restrict__ traf,
                            const float* __restrict__ W_link,
                            const float* __restrict__ W_path,
                            const float* __restrict__ a_vec,
                            float* __restrict__ link_state,
                            float* __restrict__ path_state,
                            float* __restrict__ wl, float* __restrict__ wp) {
  int tid = blockIdx.x * blockDim.x + threadIdx.x;
  int stride = gridDim.x * blockDim.x;
  for (int idx = tid; idx < NLINKS * DD; idx += stride)
    link_state[idx] = ((idx & 31) == 0) ? cap[idx >> 5] : 0.0f;
  for (int idx = tid; idx < NPATHS * DD; idx += stride)
    path_state[idx] = ((idx & 31) == 0) ? traf[idx >> 5] : 0.0f;
  if (tid < 32) {
    float s = 0.0f;
    for (int j = 0; j < 32; ++j) s += W_link[j * 32 + tid] * a_vec[j];
    wl[tid] = s;
  } else if (tid < 64) {
    int d = tid - 32;
    float s = 0.0f;
    for (int j = 0; j < 32; ++j) s += W_path[j * 32 + d] * a_vec[32 + j];
    wp[d] = s;
  }
}

__global__ void zero_kernel(float* __restrict__ msg, float* __restrict__ Ssum) {
  int tid = blockIdx.x * blockDim.x + threadIdx.x;
  int stride = gridDim.x * blockDim.x;
  for (int idx = tid; idx < NLINKS * DD; idx += stride) msg[idx] = 0.0f;
  if (tid == 0) *Ssum = 0.0f;
}

// ---------------------------------------------------------------- path RNN + attention + scatter

__global__ __launch_bounds__(32 * NW_P) void path_rnn_kernel(
    const float* __restrict__ link_state, float* __restrict__ path_state,
    const float* __restrict__ pK, const float* __restrict__ pR,
    const float* __restrict__ pb, const float* __restrict__ wl,
    const float* __restrict__ wp, const int* __restrict__ links,
    float* __restrict__ msg, float* __restrict__ Ssum) {
  __shared__ float stage[NW_P][16][33];          // A-layout staging (h, r*h, h_final)
  __shared__ float mh[NW_P][PLEN][16][DD];       // per-step outputs m
  __shared__ float elds[NW_P][PLEN][16];         // un-normalized attention e
  __shared__ int   lkl[NW_P][16][PLEN];          // link indices for this tile

  const int lane = threadIdx.x & 31;
  const int w    = threadIdx.x >> 5;
  const int m    = lane & 15;
  const int half = lane >> 4;
  const int p0   = (blockIdx.x * NW_P + w) * 16;

  for (int i = lane; i < 16 * PLEN; i += 32)
    lkl[w][i >> 3][i & 7] = links[(p0 + (i >> 3)) * PLEN + (i & 7)];

  v16h bK[6], bR[6];
#pragma unroll
  for (int g = 0; g < 6; ++g) {
    bK[g] = load_bfrag(pK, g * 16, m, half);
    bR[g] = load_bfrag(pR, g * 16, m, half);
  }
  const float bz0 = pb[m],      bz1 = pb[16 + m];
  const float br0 = pb[32 + m], br1 = pb[48 + m];
  const float bh0 = pb[64 + m], bh1 = pb[80 + m];

  float wlh[16];
#pragma unroll
  for (int i = 0; i < 8; ++i) {
    wlh[i]     = wl[(half << 3) + i];
    wlh[8 + i] = wl[16 + (half << 3) + i];
  }

  // recurrent state in C/D layout: h0 -> cols [0,16), h1 -> cols [16,32)
  v8f h0, h1;
#pragma unroll
  for (int q = 0; q < 8; ++q) {
    int p = p0 + q + 8 * half;
    h0[q] = path_state[p * DD + m];
    h1[q] = path_state[p * DD + 16 + m];
  }

  float az[PLEN];
  asm volatile("s_wait_dscnt 0x0" ::: "memory");   // lkl visible to whole wave

  for (int t = 0; t < PLEN; ++t) {
    // gather x = link_state[links[p,t]] directly into A layout; fused dot with w_l
    const float* xr = link_state + (size_t)lkl[w][m][t] * DD;
    v16h ax;
    float part = 0.0f;
#pragma unroll
    for (int i = 0; i < 8; ++i) {
      float v0 = xr[(half << 3) + i];
      float v1 = xr[16 + (half << 3) + i];
      ax[i] = (_Float16)v0;
      ax[8 + i] = (_Float16)v1;
      part += v0 * wlh[i] + v1 * wlh[8 + i];
    }
    part += __shfl_xor(part, 16);   // combine the two K-halves
    az[t] = part;

    // h: C layout -> LDS -> A layout
#pragma unroll
    for (int q = 0; q < 8; ++q) {
      stage[w][q + 8 * half][m]      = h0[q];
      stage[w][q + 8 * half][16 + m] = h1[q];
    }
    asm volatile("s_wait_dscnt 0x0" ::: "memory");
    v16h ah = load_afrag_lds(&stage[w][0][0], m, half);

    v8f zz0 = {}, zz1 = {}, gr0 = {}, gr1 = {}, cc0 = {}, cc1 = {};
    zz0 = wmma16(ax, bK[0], zz0); zz1 = wmma16(ax, bK[1], zz1);
    gr0 = wmma16(ax, bK[2], gr0); gr1 = wmma16(ax, bK[3], gr1);
    cc0 = wmma16(ax, bK[4], cc0); cc1 = wmma16(ax, bK[5], cc1);
    zz0 = wmma16(ah, bR[0], zz0); zz1 = wmma16(ah, bR[1], zz1);
    gr0 = wmma16(ah, bR[2], gr0); gr1 = wmma16(ah, bR[3], gr1);

    v8f z0, z1, rh0, rh1;
#pragma unroll
    for (int q = 0; q < 8; ++q) {
      z0[q] = fast_sigmoid(zz0[q] + bz0);
      z1[q] = fast_sigmoid(zz1[q] + bz1);
      float rg0 = fast_sigmoid(gr0[q] + br0);
      float rg1 = fast_sigmoid(gr1[q] + br1);
      rh0[q] = rg0 * h0[q];
      rh1[q] = rg1 * h1[q];
    }
#pragma unroll
    for (int q = 0; q < 8; ++q) {
      stage[w][q + 8 * half][m]      = rh0[q];
      stage[w][q + 8 * half][16 + m] = rh1[q];
    }
    asm volatile("s_wait_dscnt 0x0" ::: "memory");
    v16h arh = load_afrag_lds(&stage[w][0][0], m, half);
    cc0 = wmma16(arh, bR[4], cc0);
    cc1 = wmma16(arh, bR[5], cc1);

#pragma unroll
    for (int q = 0; q < 8; ++q) {
      float hh0v = fast_tanh(cc0[q] + bh0);
      float hh1v = fast_tanh(cc1[q] + bh1);
      h0[q] = hh0v + z0[q] * (h0[q] - hh0v);   // z*h + (1-z)*hh
      h1[q] = hh1v + z1[q] * (h1[q] - hh1v);
      mh[w][t][q + 8 * half][m]      = h0[q];
      mh[w][t][q + 8 * half][16 + m] = h1[q];
    }
  }

  // attention: az_h = h_final . w_p (via staged h_final)
#pragma unroll
  for (int q = 0; q < 8; ++q) {
    stage[w][q + 8 * half][m]      = h0[q];
    stage[w][q + 8 * half][16 + m] = h1[q];
  }
  asm volatile("s_wait_dscnt 0x0" ::: "memory");
  float azh = 0.0f;
#pragma unroll
  for (int c = 0; c < DD; ++c) azh += stage[w][m][c] * wp[c];

  float sloc = 0.0f;
#pragma unroll
  for (int t = 0; t < PLEN; ++t) {
    float e = az[t] + azh;
    e = e > 0.0f ? e : 0.0f;
    if (half == 0) {           // each (path,t) counted once
      elds[w][t][m] = e;
      sloc += e * e;
    }
  }
  asm volatile("s_wait_dscnt 0x0" ::: "memory");
#pragma unroll
  for (int off = 16; off > 0; off >>= 1) sloc += __shfl_xor(sloc, off);
  if (lane == 0) atomicAdd(Ssum, sloc);

  // write back updated path state
#pragma unroll
  for (int q = 0; q < 8; ++q) {
    int p = p0 + q + 8 * half;
    path_state[p * DD + m]      = h0[q];
    path_state[p * DD + 16 + m] = h1[q];
  }

  // scatter un-normalized e * m into msg: one coalesced 32-lane atomic row per edge
  for (int eidx = 0; eidx < 16 * PLEN; ++eidx) {
    int pl = eidx >> 3, t = eidx & 7;
    float e = elds[w][t][pl];
    if (e > 0.0f) {
      int lnk = lkl[w][pl][t];
      atomicAdd(&msg[(size_t)lnk * DD + lane], e * mh[w][t][pl][lane]);
    }
  }
}

// ---------------------------------------------------------------- link GRU

__global__ __launch_bounds__(32 * NW_L) void link_gru_kernel(
    float* __restrict__ link_state, const float* __restrict__ msg,
    const float* __restrict__ eK, const float* __restrict__ eR,
    const float* __restrict__ ebv, const float* __restrict__ Ssum) {
  __shared__ float stage[NW_L][16][33];
  const int lane = threadIdx.x & 31;
  const int w    = threadIdx.x >> 5;
  const int m    = lane & 15;
  const int half = lane >> 4;
  const int l0   = (blockIdx.x * NW_L + w) * 16;

  const float inv = rsqrtf(fmaxf(*Ssum, 1e-12f));  // fold global l2_normalize here

  v16h bK[6], bR[6];
#pragma unroll
  for (int g = 0; g < 6; ++g) {
    bK[g] = load_bfrag(eK, g * 16, m, half);
    bR[g] = load_bfrag(eR, g * 16, m, half);
  }
  const float bz0 = ebv[m],      bz1 = ebv[16 + m];
  const float br0 = ebv[32 + m], br1 = ebv[48 + m];
  const float bh0 = ebv[64 + m], bh1 = ebv[80 + m];

  v16h ax = load_afrag_g(msg + (size_t)(l0 + m) * DD, half, inv);
  v16h ah = load_afrag_g(link_state + (size_t)(l0 + m) * DD, half, 1.0f);

  v8f h0, h1;
#pragma unroll
  for (int q = 0; q < 8; ++q) {
    int l = l0 + q + 8 * half;
    h0[q] = link_state[l * DD + m];
    h1[q] = link_state[l * DD + 16 + m];
  }

  v8f zz0 = {}, zz1 = {}, gr0 = {}, gr1 = {}, cc0 = {}, cc1 = {};
  zz0 = wmma16(ax, bK[0], zz0); zz1 = wmma16(ax, bK[1], zz1);
  gr0 = wmma16(ax, bK[2], gr0); gr1 = wmma16(ax, bK[3], gr1);
  cc0 = wmma16(ax, bK[4], cc0); cc1 = wmma16(ax, bK[5], cc1);
  zz0 = wmma16(ah, bR[0], zz0); zz1 = wmma16(ah, bR[1], zz1);
  gr0 = wmma16(ah, bR[2], gr0); gr1 = wmma16(ah, bR[3], gr1);

  v8f z0, z1, rh0, rh1;
#pragma unroll
  for (int q = 0; q < 8; ++q) {
    z0[q] = fast_sigmoid(zz0[q] + bz0);
    z1[q] = fast_sigmoid(zz1[q] + bz1);
    float rg0 = fast_sigmoid(gr0[q] + br0);
    float rg1 = fast_sigmoid(gr1[q] + br1);
    rh0[q] = rg0 * h0[q];
    rh1[q] = rg1 * h1[q];
  }
#pragma unroll
  for (int q = 0; q < 8; ++q) {
    stage[w][q + 8 * half][m]      = rh0[q];
    stage[w][q + 8 * half][16 + m] = rh1[q];
  }
  asm volatile("s_wait_dscnt 0x0" ::: "memory");
  v16h arh = load_afrag_lds(&stage[w][0][0], m, half);
  cc0 = wmma16(arh, bR[4], cc0);
  cc1 = wmma16(arh, bR[5], cc1);

#pragma unroll
  for (int q = 0; q < 8; ++q) {
    float hh0v = fast_tanh(cc0[q] + bh0);
    float hh1v = fast_tanh(cc1[q] + bh1);
    h0[q] = hh0v + z0[q] * (h0[q] - hh0v);
    h1[q] = hh1v + z1[q] * (h1[q] - hh1v);
    int l = l0 + q + 8 * half;
    link_state[l * DD + m]      = h0[q];
    link_state[l * DD + 16 + m] = h1[q];
  }
}

// ---------------------------------------------------------------- readout MLP

__global__ void readout_kernel(const float* __restrict__ ps,
                               const float* __restrict__ r1w, const float* __restrict__ r1b,
                               const float* __restrict__ r2w, const float* __restrict__ r2b,
                               const float* __restrict__ finw, const float* __restrict__ finb,
                               float* __restrict__ out) {
  int p = blockIdx.x * blockDim.x + threadIdx.x;
  if (p >= NPATHS) return;
  float h[DD];
#pragma unroll
  for (int c = 0; c < DD; ++c) h[c] = ps[p * DD + c];
  float r1[8];
#pragma unroll
  for (int j = 0; j < 8; ++j) {
    float a = r1b[j];
#pragma unroll
    for (int c = 0; c < DD; ++c) a += h[c] * r1w[c * 8 + j];
    r1[j] = seluf_(a);
  }
  float r2[8];
#pragma unroll
  for (int j = 0; j < 8; ++j) {
    float a = r2b[j];
#pragma unroll
    for (int c = 0; c < 8; ++c) a += r1[c] * r2w[c * 8 + j];
    r2[j] = seluf_(a);
  }
  float o = finb[0];
#pragma unroll
  for (int j = 0; j < 8; ++j) o += r2[j] * finw[j];
#pragma unroll
  for (int c = 0; c < DD; ++c) o += h[c] * finw[8 + c];
  out[p] = o;
}

// ---------------------------------------------------------------- launch

extern "C" void kernel_launch(void* const* d_in, const int* in_sizes, int n_in,
                              void* d_out, int out_size, void* d_ws, size_t ws_size,
                              hipStream_t stream) {
  const float* capacities = (const float*)d_in[0];
  const float* traffic    = (const float*)d_in[1];
  const float* W_link     = (const float*)d_in[2];
  const float* W_path     = (const float*)d_in[3];
  const float* a_vec      = (const float*)d_in[4];
  const float* pK         = (const float*)d_in[5];
  const float* pR         = (const float*)d_in[6];
  const float* pb         = (const float*)d_in[7];
  const float* eK         = (const float*)d_in[8];
  const float* eR         = (const float*)d_in[9];
  const float* eb         = (const float*)d_in[10];
  const float* r1w        = (const float*)d_in[11];
  const float* r1b        = (const float*)d_in[12];
  const float* r2w        = (const float*)d_in[13];
  const float* r2b        = (const float*)d_in[14];
  const float* finw       = (const float*)d_in[15];
  const float* finb       = (const float*)d_in[16];
  const int*   links      = (const int*)d_in[17];
  // d_in[18] (paths) / d_in[19] (seqs) are structurally paths=e/8, seqs=e%8

  float* ws         = (float*)d_ws;
  float* path_state = ws;
  float* link_state = path_state + (size_t)NPATHS * DD;
  float* msg        = link_state + (size_t)NLINKS * DD;
  float* wl         = msg + (size_t)NLINKS * DD;
  float* wp         = wl + 32;
  float* Ssum       = wp + 32;

  float* out = (float*)d_out;

  init_kernel<<<1024, 256, 0, stream>>>(capacities, traffic, W_link, W_path, a_vec,
                                        link_state, path_state, wl, wp);
  for (int it = 0; it < TITER; ++it) {
    zero_kernel<<<256, 256, 0, stream>>>(msg, Ssum);
    path_rnn_kernel<<<NPATHS / (16 * NW_P), 32 * NW_P, 0, stream>>>(
        link_state, path_state, pK, pR, pb, wl, wp, links, msg, Ssum);
    link_gru_kernel<<<NLINKS / (16 * NW_L), 32 * NW_L, 0, stream>>>(
        link_state, msg, eK, eR, eb, Ssum);
  }
  readout_kernel<<<(NPATHS + 255) / 256, 256, 0, stream>>>(
      path_state, r1w, r1b, r2w, r2b, finw, finb, out);
}
